// GNNModel_2259152798069
// MI455X (gfx1250) — matrix operand order repaired and checked by reading, not compiled, not run on hardware
//
#include <hip/hip_runtime.h>
#include <hip/hip_bf16.h>

typedef __attribute__((ext_vector_type(16))) _Float16 v16h;
typedef __attribute__((ext_vector_type(8)))  _Float16 v8h;
typedef __attribute__((ext_vector_type(4)))  _Float16 v4h;
typedef __attribute__((ext_vector_type(8)))  float    v8f;

#define N_NODES 16384
#define N_EDGES 65536
#define N_SESS  512
#define SESS_L  32
#define HID     64
#define NHEADS  16
#define NINT    3
#define VOCAB   32000
#define DDIM    192
#define LRELU   0.2f

// ---------------- elementwise / graph kernels ----------------

__global__ void k_embed_gate(const int* __restrict__ x, const float* __restrict__ emb,
                             const float* __restrict__ wint, float* __restrict__ e,
                             float* __restrict__ g)
{
    int n = blockIdx.x * blockDim.x + threadIdx.x;
    if (n >= N_NODES) return;
    const float* row = emb + (size_t)x[n] * HID;
    float l0 = 0.f, l1 = 0.f, l2 = 0.f;
    for (int h = 0; h < HID; h += 4) {
        float4 v = *(const float4*)(row + h);
        *(float4*)(e + (size_t)n * HID + h) = v;
        l0 += v.x * wint[(h+0)*3+0] + v.y * wint[(h+1)*3+0] + v.z * wint[(h+2)*3+0] + v.w * wint[(h+3)*3+0];
        l1 += v.x * wint[(h+0)*3+1] + v.y * wint[(h+1)*3+1] + v.z * wint[(h+2)*3+1] + v.w * wint[(h+3)*3+1];
        l2 += v.x * wint[(h+0)*3+2] + v.y * wint[(h+1)*3+2] + v.z * wint[(h+2)*3+2] + v.w * wint[(h+3)*3+2];
    }
    l0 *= 10.f; l1 *= 10.f; l2 *= 10.f;   // /0.1 temperature
    float m  = fmaxf(l0, fmaxf(l1, l2));
    float e0 = __expf(l0 - m), e1 = __expf(l1 - m), e2 = __expf(l2 - m);
    float inv = 1.f / (e0 + e1 + e2);
    g[n * 3 + 0] = e0 * inv; g[n * 3 + 1] = e1 * inv; g[n * 3 + 2] = e2 * inv;
}

__global__ void k_fill(float* __restrict__ p, float v, size_t cnt)
{
    size_t i = (size_t)blockIdx.x * blockDim.x + threadIdx.x;
    if (i < cnt) p[i] = v;
}

__device__ inline void atomicMaxF(float* addr, float val)
{
    unsigned* ua = (unsigned*)addr;
    unsigned old = *ua;
    while (__uint_as_float(old) < val) {
        unsigned assumed = old;
        old = atomicCAS(ua, assumed, __float_as_uint(val));
        if (old == assumed) break;
    }
}

// heads = 1 << hshift
__global__ void k_attn(const float* __restrict__ h, const float* __restrict__ a_s,
                       const float* __restrict__ a_d, float* __restrict__ asrc,
                       float* __restrict__ adst, int hshift)
{
    int t = blockIdx.x * blockDim.x + threadIdx.x;
    if (t >= (N_NODES << hshift)) return;
    int hd = t & ((1 << hshift) - 1);
    const float* hp = h + (size_t)t * HID;          // t == n*heads + hd
    const float* sp = a_s + hd * HID;
    const float* dp = a_d + hd * HID;
    float s = 0.f, d = 0.f;
    for (int o = 0; o < HID; o += 4) {
        float4 v = *(const float4*)(hp + o);
        float4 a = *(const float4*)(sp + o);
        float4 b = *(const float4*)(dp + o);
        s += v.x*a.x + v.y*a.y + v.z*a.z + v.w*a.w;
        d += v.x*b.x + v.y*b.y + v.z*b.z + v.w*b.w;
    }
    asrc[t] = s; adst[t] = d;
}

__device__ inline bool edge_sd(const int* ei, const float* g, int interest,
                               int eid, int& s, int& d)
{
    if (eid < N_EDGES) {
        s = ei[eid]; d = ei[N_EDGES + eid];
        return (g[s * 3 + interest] > (1.f / 3.f)) && (g[d * 3 + interest] > (1.f / 3.f));
    }
    s = d = eid - N_EDGES;      // self loop, always valid
    return true;
}

__global__ void k_edge_max(const int* __restrict__ ei, const float* __restrict__ g, int interest,
                           const float* __restrict__ asrc, const float* __restrict__ adst,
                           float* __restrict__ maxlog, int hshift)
{
    int t = blockIdx.x * blockDim.x + threadIdx.x;
    if (t >= ((N_EDGES + N_NODES) << hshift)) return;
    int eid = t >> hshift, hd = t & ((1 << hshift) - 1), s, d;
    if (!edge_sd(ei, g, interest, eid, s, d)) return;
    float l = asrc[(s << hshift) + hd] + adst[(d << hshift) + hd];
    l = (l >= 0.f) ? l : LRELU * l;
    atomicMaxF(&maxlog[(d << hshift) + hd], l);
}

__global__ void k_edge_sum(const int* __restrict__ ei, const float* __restrict__ g, int interest,
                           const float* __restrict__ asrc, const float* __restrict__ adst,
                           const float* __restrict__ maxlog, const float* __restrict__ h,
                           float* __restrict__ denom, float* __restrict__ num, int hshift)
{
    int t = blockIdx.x * blockDim.x + threadIdx.x;
    if (t >= ((N_EDGES + N_NODES) << hshift)) return;
    int eid = t >> hshift, hd = t & ((1 << hshift) - 1), s, d;
    if (!edge_sd(ei, g, interest, eid, s, d)) return;
    float l = asrc[(s << hshift) + hd] + adst[(d << hshift) + hd];
    l = (l >= 0.f) ? l : LRELU * l;
    float p = __expf(l - maxlog[(d << hshift) + hd]);
    atomicAdd(&denom[(d << hshift) + hd], p);
    const float* hp = h   + ((size_t)(s << hshift) + hd) * HID;
    float*       np = num + ((size_t)(d << hshift) + hd) * HID;
    for (int o = 0; o < HID; ++o) atomicAdd(&np[o], p * hp[o]);
}

__global__ void k_finalize(const float* __restrict__ num, const float* __restrict__ den,
                           const float* __restrict__ bias, float* __restrict__ outp,
                           int hshift, int outStride, int outOfs, int relu)
{
    int t = blockIdx.x * blockDim.x + threadIdx.x;
    if (t >= (N_NODES << hshift) * HID) return;
    int n = t >> (hshift + 6), rem = t & (((1 << hshift) * HID) - 1);
    float v = num[t] / den[t >> 6] + bias[rem];
    if (relu) v = fmaxf(v, 0.f);
    outp[(size_t)n * outStride + outOfs + rem] = v;
}

__global__ void k_vn(const float* __restrict__ sess, float* __restrict__ vn)
{
    int t = blockIdx.x * blockDim.x + threadIdx.x;
    if (t >= N_SESS * DDIM) return;
    int b = t / DDIM, d = t - b * DDIM;
    vn[t] = sess[((size_t)b * SESS_L + (SESS_L - 1)) * DDIM + d];
}

__global__ __launch_bounds__(256)
void k_alpha_sg(const float* __restrict__ q1, const float* __restrict__ q2,
                const float* __restrict__ qw, const float* __restrict__ qb,
                const float* __restrict__ sess, const int* __restrict__ batch,
                float* __restrict__ sg)
{
    __shared__ float red[256];
    __shared__ float alphaS;
    int n = blockIdx.x, tid = threadIdx.x;
    float p = 0.f;
    if (tid < DDIM) {
        float s = q1[(size_t)n * DDIM + tid] + q2[(size_t)n * DDIM + tid];
        s = 1.f / (1.f + __expf(-s));
        p = s * qw[tid];
    }
    red[tid] = p; __syncthreads();
    for (int st = 128; st > 0; st >>= 1) { if (tid < st) red[tid] += red[tid + st]; __syncthreads(); }
    if (tid == 0) alphaS = red[0] + qb[0];
    __syncthreads();
    if (tid < DDIM)
        atomicAdd(&sg[(size_t)batch[n] * DDIM + tid], alphaS * sess[(size_t)n * DDIM + tid]);
}

__global__ void k_cat(const float* __restrict__ vn, const float* __restrict__ sg,
                      float* __restrict__ cat)
{
    int t = blockIdx.x * blockDim.x + threadIdx.x;
    if (t >= N_SESS * 2 * DDIM) return;
    int b = t / (2 * DDIM), d = t - b * 2 * DDIM;
    cat[t] = (d < DDIM) ? vn[(size_t)b * DDIM + d] : sg[(size_t)b * DDIM + (d - DDIM)];
}

// ---------------- WMMA GEMM: C[M,Nc] = A[M,Kd] (opt. row-gated) @ W[Kd,Nc] + bias ----------------
// 4 waves/block, 16x64 block tile, K-step 64 (2 WMMAs per stage).
// Tiles staged in LDS as f16, B pre-transposed to the register layout, so each WMMA
// operand is filled by two contiguous ds_load_b128 with no per-use conversion.
// rowShift: logical row -> A row (0 normally, 5 for v_n[batch]).  Kd must be a multiple of 64.

__global__ __launch_bounds__(128)
void k_wmma_gemm(const float* __restrict__ A, const float* __restrict__ Wm,
                 const float* __restrict__ bias, float* __restrict__ C,
                 int Kd, int Nc,
                 const float* __restrict__ gate, int gofs, int rowShift)
{
    __shared__ _Float16 As16[16 * 64];   // [row][k]      (2 KB)
    __shared__ _Float16 Bsw[64 * 64];    // [col][k]      (8 KB) transposed W tile
    const int tid  = threadIdx.x;
    const int lane = tid & 31;
    const int wave = tid >> 5;
    const int half = lane >> 4;       // wave32: lanes 0-15 / 16-31
    const int r    = lane & 15;
    const int row0 = blockIdx.y * 16;
    const int colBase = blockIdx.x * 64;
    const int cc   = wave * 16 + r;   // column within block tile
    const int col  = colBase + cc;

    v8f acc = {};
    for (int k0 = 0; k0 < Kd; k0 += 64) {
        // ---- stage A tile (16x64) -> f16, convert once; uniform branch on gate ----
        if (gate) {
            for (int idx = tid; idx < 256; idx += 128) {
                int rr = idx >> 4, kk = (idx & 15) << 2;
                int lrow = row0 + rr;
                float4 v = *(const float4*)&A[(size_t)(lrow >> rowShift) * Kd + k0 + kk];
                float sc = gate[lrow * 3 + gofs];
                *(v4h*)&As16[rr * 64 + kk] =
                    (v4h){(_Float16)(v.x*sc), (_Float16)(v.y*sc), (_Float16)(v.z*sc), (_Float16)(v.w*sc)};
            }
        } else {
            for (int idx = tid; idx < 256; idx += 128) {
                int rr = idx >> 4, kk = (idx & 15) << 2;
                float4 v = *(const float4*)&A[(size_t)((row0 + rr) >> rowShift) * Kd + k0 + kk];
                *(v4h*)&As16[rr * 64 + kk] =
                    (v4h){(_Float16)v.x, (_Float16)v.y, (_Float16)v.z, (_Float16)v.w};
            }
        }
        // ---- stage B tile (64K x 64col) transposed -> Bsw[col][k], f16 ----
        // idx -> (col c, k-quad kq); 4 global loads down the column are coalesced across c.
        for (int idx = tid; idx < 1024; idx += 128) {
            int c = idx & 63, kk = (idx >> 6) << 2;
            const float* wp = &Wm[(size_t)(k0 + kk) * Nc + colBase + c];
            *(v4h*)&Bsw[c * 64 + kk] =
                (v4h){(_Float16)wp[0], (_Float16)wp[(size_t)Nc],
                      (_Float16)wp[(size_t)2 * Nc], (_Float16)wp[(size_t)3 * Nc]};
        }
        __syncthreads();
        if (k0 + 64 < Kd)
            __builtin_prefetch(&Wm[(size_t)(k0 + 64) * Nc + colBase + (tid & 63)], 0, 1);
        #pragma unroll
        for (int ks = 0; ks < 64; ks += 32) {
            // A 16x32 f16 layout: lane(half,r) needs runs [ks+8h, +8) and [ks+16+8h, +8)
            union { v16h v; v8h h[2]; } au, bu;
            au.h[0] = *(const v8h*)&As16[r * 64 + ks + 8 * half];
            au.h[1] = *(const v8h*)&As16[r * 64 + ks + 16 + 8 * half];
            // B 32x16 f16 layout: lane(half,n) needs runs [ks+8h, +8) and [ks+16+8h, +8) of its column
            bu.h[0] = *(const v8h*)&Bsw[cc * 64 + ks + 8 * half];
            bu.h[1] = *(const v8h*)&Bsw[cc * 64 + ks + 16 + 8 * half];
            acc = __builtin_amdgcn_wmma_f32_16x16x32_f16(false, au.v, false, bu.v,
                                                         (short)0, acc, false, false);
        }
        __syncthreads();
    }
    float bv = bias ? bias[col] : 0.f;
    #pragma unroll
    for (int rr = 0; rr < 8; ++rr) {       // D layout: VGPR rr -> M = rr + 8*half
        int m = row0 + rr + 8 * half;
        C[(size_t)m * Nc + col] = acc[rr] + bv;
    }
}

// ---------------- scoring: out[b,v] = max_k  s_h[b*3+k,:] . emb[v,:] ----------------
// block = 4 waves; 48 s_h rows x 64 vocab cols; max over K fused on the LDS tile.

__global__ __launch_bounds__(128)
void k_score(const float* __restrict__ sh, const float* __restrict__ emb,
             float* __restrict__ out)
{
    __shared__ float zbuf[48 * 64];
    const int tid  = threadIdx.x;
    const int lane = tid & 31;
    const int wave = tid >> 5;           // column tile
    const int half = lane >> 4;
    const int r    = lane & 15;
    const int v    = blockIdx.x * 64 + wave * 16 + r;
    const int s0   = blockIdx.y * 48;

    __builtin_amdgcn_s_wait_tensorcnt(0);
    #pragma unroll
    for (int rt = 0; rt < 3; ++rt) {
        v8f acc = {};
        #pragma unroll
        for (int k0 = 0; k0 < HID; k0 += 32) {
            v16h a, b;
            // A: lanes hold rows of s_h; this lane needs 16 contiguous K starting at 16*half
            const float* arow = sh + (size_t)(s0 + rt * 16 + r) * HID + k0;
            float4 aL0 = *(const float4*)(arow + 16 * half);
            float4 aL1 = *(const float4*)(arow + 16 * half + 4);
            float4 aH0 = *(const float4*)(arow + 16 * half + 8);
            float4 aH1 = *(const float4*)(arow + 16 * half + 12);
            a[0]=(_Float16)aL0.x; a[1]=(_Float16)aL0.y; a[2]=(_Float16)aL0.z; a[3]=(_Float16)aL0.w;
            a[4]=(_Float16)aL1.x; a[5]=(_Float16)aL1.y; a[6]=(_Float16)aL1.z; a[7]=(_Float16)aL1.w;
            a[8]=(_Float16)aH0.x; a[9]=(_Float16)aH0.y; a[10]=(_Float16)aH0.z; a[11]=(_Float16)aH0.w;
            a[12]=(_Float16)aH1.x; a[13]=(_Float16)aH1.y; a[14]=(_Float16)aH1.z; a[15]=(_Float16)aH1.w;
            // B: column = emb row v; this lane needs 16 contiguous floats starting at 16*half
            const float* brow = emb + (size_t)v * HID + k0 + 16 * half;
            #pragma unroll
            for (int q = 0; q < 4; ++q) {
                float4 bv4 = *(const float4*)(brow + 4 * q);
                b[4*q+0] = (_Float16)bv4.x; b[4*q+1] = (_Float16)bv4.y;
                b[4*q+2] = (_Float16)bv4.z; b[4*q+3] = (_Float16)bv4.w;
            }
            acc = __builtin_amdgcn_wmma_f32_16x16x32_f16(false, a, false, b,
                                                         (short)0, acc, false, false);
        }
        #pragma unroll
        for (int rr = 0; rr < 8; ++rr)
            zbuf[(rt * 16 + rr + 8 * half) * 64 + wave * 16 + r] = acc[rr];
    }
    __syncthreads();
    for (int idx = tid; idx < 16 * 64; idx += 128) {
        int s = idx >> 6, c = idx & 63;
        float m = zbuf[(s * 3 + 0) * 64 + c];
        m = fmaxf(m, zbuf[(s * 3 + 1) * 64 + c]);
        m = fmaxf(m, zbuf[(s * 3 + 2) * 64 + c]);
        out[(size_t)(blockIdx.y * 16 + s) * VOCAB + blockIdx.x * 64 + c] = m;
    }
}

// ---------------- host launcher ----------------

static inline int cdiv(long long a, int b) { return (int)((a + b - 1) / b); }

extern "C" void kernel_launch(void* const* d_in, const int* in_sizes, int n_in,
                              void* d_out, int out_size, void* d_ws, size_t ws_size,
                              hipStream_t stream)
{
    const int*   x     = (const int*)d_in[0];
    const int*   ei    = (const int*)d_in[1];
    const int*   batch = (const int*)d_in[2];
    const float* emb   = (const float*)d_in[3];
    const float* wint  = (const float*)d_in[4];
    const float* g1W   = (const float*)d_in[5];
    const float* g1as  = (const float*)d_in[6];
    const float* g1ad  = (const float*)d_in[7];
    const float* g1b   = (const float*)d_in[8];
    const float* g2W   = (const float*)d_in[9];
    const float* g2as  = (const float*)d_in[10];
    const float* g2ad  = (const float*)d_in[11];
    const float* g2b   = (const float*)d_in[12];
    const float* W1w   = (const float*)d_in[13];
    const float* W1b   = (const float*)d_in[14];
    const float* W2w   = (const float*)d_in[15];
    const float* W2b   = (const float*)d_in[16];
    const float* qw    = (const float*)d_in[17];
    const float* qb    = (const float*)d_in[18];
    const float* W3w   = (const float*)d_in[19];
    const float* W3b   = (const float*)d_in[20];
    float* out = (float*)d_out;

    // workspace carve-up (floats)
    float* ws = (float*)d_ws;
    size_t off = 0;
    auto alloc = [&](size_t n) { float* p = ws + off; off += n; return p; };
    float* e      = alloc((size_t)N_NODES * HID);
    float* g      = alloc((size_t)N_NODES * NINT);
    float* asrc   = alloc((size_t)N_NODES * NHEADS);
    float* adst   = alloc((size_t)N_NODES * NHEADS);
    float* maxlog = alloc((size_t)N_NODES * NHEADS);
    float* denom  = alloc((size_t)N_NODES * NHEADS);
    float* h1lin  = alloc((size_t)N_NODES * NHEADS * HID);
    float* num1   = alloc((size_t)N_NODES * NHEADS * HID);   // also h1 final (in-place)
    float* h2lin  = alloc((size_t)N_NODES * HID);
    float* num2   = alloc((size_t)N_NODES * HID);
    float* sess   = alloc((size_t)N_NODES * DDIM);
    float* q1     = alloc((size_t)N_NODES * DDIM);
    float* q2     = alloc((size_t)N_NODES * DDIM);
    float* vn     = alloc((size_t)N_SESS * DDIM);
    float* sg     = alloc((size_t)N_SESS * DDIM);
    float* catb   = alloc((size_t)N_SESS * 2 * DDIM);
    float* s_h    = alloc((size_t)N_SESS * DDIM);

    // 1) embed + interest gate softmax
    k_embed_gate<<<cdiv(N_NODES, 256), 256, 0, stream>>>(x, emb, wint, e, g);

    const int EV1 = (N_EDGES + N_NODES) * NHEADS;
    const int EV2 = (N_EDGES + N_NODES);

    for (int i = 0; i < NINT; ++i) {
        // ---- GAT layer 1 (16 heads, hshift=4) ----
        k_fill<<<cdiv((size_t)N_NODES * NHEADS, 256), 256, 0, stream>>>(maxlog, -1e30f, (size_t)N_NODES * NHEADS);
        k_fill<<<cdiv((size_t)N_NODES * NHEADS, 256), 256, 0, stream>>>(denom, 0.f, (size_t)N_NODES * NHEADS);
        k_fill<<<cdiv((size_t)N_NODES * NHEADS * HID, 256), 256, 0, stream>>>(num1, 0.f, (size_t)N_NODES * NHEADS * HID);

        k_wmma_gemm<<<dim3((NHEADS * HID) / 64, N_NODES / 16), 128, 0, stream>>>(
            e, g1W + (size_t)i * HID * NHEADS * HID, nullptr, h1lin,
            HID, NHEADS * HID, g, i, 0);

        k_attn<<<cdiv(N_NODES * NHEADS, 256), 256, 0, stream>>>(
            h1lin, g1as + (size_t)i * NHEADS * HID, g1ad + (size_t)i * NHEADS * HID,
            asrc, adst, 4);
        k_edge_max<<<cdiv(EV1, 256), 256, 0, stream>>>(ei, g, i, asrc, adst, maxlog, 4);
        k_edge_sum<<<cdiv(EV1, 256), 256, 0, stream>>>(ei, g, i, asrc, adst, maxlog, h1lin,
                                                       denom, num1, 4);
        k_finalize<<<cdiv((long long)N_NODES * NHEADS * HID, 256), 256, 0, stream>>>(
            num1, denom, g1b + (size_t)i * NHEADS * HID, num1,
            4, NHEADS * HID, 0, /*relu=*/1);

        // ---- GAT layer 2 (1 head, hshift=0) ----
        k_fill<<<cdiv((size_t)N_NODES, 256), 256, 0, stream>>>(maxlog, -1e30f, (size_t)N_NODES);
        k_fill<<<cdiv((size_t)N_NODES, 256), 256, 0, stream>>>(denom, 0.f, (size_t)N_NODES);
        k_fill<<<cdiv((size_t)N_NODES * HID, 256), 256, 0, stream>>>(num2, 0.f, (size_t)N_NODES * HID);

        k_wmma_gemm<<<dim3(HID / 64, N_NODES / 16), 128, 0, stream>>>(
            num1, g2W + (size_t)i * NHEADS * HID * HID, nullptr, h2lin,
            NHEADS * HID, HID, nullptr, 0, 0);

        k_attn<<<cdiv(N_NODES, 256), 256, 0, stream>>>(
            h2lin, g2as + (size_t)i * HID, g2ad + (size_t)i * HID, asrc, adst, 0);
        k_edge_max<<<cdiv(EV2, 256), 256, 0, stream>>>(ei, g, i, asrc, adst, maxlog, 0);
        k_edge_sum<<<cdiv(EV2, 256), 256, 0, stream>>>(ei, g, i, asrc, adst, maxlog, h2lin,
                                                       denom, num2, 0);
        k_finalize<<<cdiv(N_NODES * HID, 256), 256, 0, stream>>>(
            num2, denom, g2b + (size_t)i * HID, sess, 0, DDIM, i * HID, /*relu=*/0);
    }

    // Embedding2Score
    k_vn<<<cdiv(N_SESS * DDIM, 256), 256, 0, stream>>>(sess, vn);
    k_wmma_gemm<<<dim3(DDIM / 64, N_NODES / 16), 128, 0, stream>>>(
        vn, W1w, W1b, q1, DDIM, DDIM, nullptr, 0, /*rowShift=*/5);
    k_wmma_gemm<<<dim3(DDIM / 64, N_NODES / 16), 128, 0, stream>>>(
        sess, W2w, W2b, q2, DDIM, DDIM, nullptr, 0, 0);
    k_fill<<<cdiv((size_t)N_SESS * DDIM, 256), 256, 0, stream>>>(sg, 0.f, (size_t)N_SESS * DDIM);
    k_alpha_sg<<<N_NODES, 256, 0, stream>>>(q1, q2, qw, qb, sess, batch, sg);
    k_cat<<<cdiv(N_SESS * 2 * DDIM, 256), 256, 0, stream>>>(vn, sg, catb);
    k_wmma_gemm<<<dim3(DDIM / 64, N_SESS / 16), 128, 0, stream>>>(
        catb, W3w, W3b, s_h, 2 * DDIM, DDIM, nullptr, 0, 0);

    // final scoring GEMM with fused max over interests
    k_score<<<dim3(VOCAB / 64, N_SESS / 16), 128, 0, stream>>>(s_h, emb, out);
}
